// SimpleVQAutoEncoder_45148696216876
// MI455X (gfx1250) — compile-verified, matrix operand order checked by reading, not run on hardware
//
#include <hip/hip_runtime.h>
#include <math.h>

// ---------------------------------------------------------------------------
// Types
// ---------------------------------------------------------------------------
typedef _Float16 half8  __attribute__((ext_vector_type(8)));
typedef _Float16 v16h   __attribute__((ext_vector_type(16)));
typedef float    v8f    __attribute__((ext_vector_type(8)));
typedef float    f32x4  __attribute__((ext_vector_type(4)));
typedef unsigned int u32x4 __attribute__((ext_vector_type(4)));
typedef int      i32x4 __attribute__((ext_vector_type(4)));
typedef int      i32x8 __attribute__((ext_vector_type(8)));

#define LDST 40   // LDS row stride in halves: 32 data + 8 pad (16 data DW + 4 pad DW)
#define OST  136  // epilogue LDS tile stride in halves (128 data + 8 pad)

#ifndef USE_TDM
#if defined(__gfx1250__) && __has_builtin(__builtin_amdgcn_tensor_load_to_lds) && \
    __has_builtin(__builtin_amdgcn_s_wait_tensorcnt)
#define USE_TDM 1
#else
#define USE_TDM 0
#endif
#endif

#if USE_TDM
// Issue one TDM 2D tile load (f16 elements) into LDS. Tensor dims are the
// *remaining* extent from the tile origin so OOB reads zero-fill the tile.
// LDS padding: after every 16 DWORDs (one 32-half row) insert 4 DWORDs,
// matching the LDST=40 row stride.
__device__ __forceinline__ void tdm_load_tile_f16(
    uint32_t lds_byte, const _Float16* gsrc, int rem_k, int rem_rows,
    int tile_rows, int stride_elems) {
  const uint64_t ga = (uint64_t)(uintptr_t)gsrc;
  u32x4 g0;
  g0[0] = 1u;                                                  // count=1
  g0[1] = lds_byte;                                            // lds_addr
  g0[2] = (uint32_t)ga;                                        // global_addr lo
  g0[3] = (uint32_t)((ga >> 32) & 0x01FFFFFFu) | (2u << 30);   // hi + type=2
  i32x8 g1;
  // data_size=1 (2B) | pad_enable | pad_interval=3 (16 DW) | pad_amount=3 (4 DW)
  g1[0] = (1 << 16) | (1 << 20) | (3 << 22) | (3 << 25);
  g1[1] = (int)(((uint32_t)rem_k & 0xFFFFu) << 16);            // tensor_dim0 lo16
  g1[2] = (int)(((uint32_t)rem_k >> 16) |
                (((uint32_t)rem_rows & 0xFFFFu) << 16));       // dim0 hi | dim1 lo
  g1[3] = (int)(((uint32_t)rem_rows >> 16) | (32u << 16));     // dim1 hi | tile_dim0=32
  g1[4] = tile_rows & 0xFFFF;                                  // tile_dim1
  g1[5] = stride_elems;                                        // dim0 stride lo32
  g1[6] = 0;
  g1[7] = 0;
  const i32x4 z4 = {0, 0, 0, 0};
#if __clang_major__ >= 23
  const i32x8 z8 = {0, 0, 0, 0, 0, 0, 0, 0};
  __builtin_amdgcn_tensor_load_to_lds(g0, g1, z4, z4, z8, 0);
#else
  __builtin_amdgcn_tensor_load_to_lds(g0, g1, z4, z4, 0);
#endif
}
#endif

// End-of-stage sync. TENSORcnt is per-wave (0 on non-issuing waves), so the
// wait is unconditional -> no exec-mask dance around it.
__device__ __forceinline__ void pipe_barrier() {
#if USE_TDM
  __builtin_amdgcn_s_wait_tensorcnt(0);
#endif
  __syncthreads();
}

// 16x32 f16 fragment from an LDS tile (CDNA5 ISA 7.12.2 16-bit A layout;
// B uses the same pattern with lane->column since tiles are staged [N][K]).
__device__ __forceinline__ v16h ld_frag(const _Float16* tile) {
  const int lane = threadIdx.x & 31;
  const _Float16* p = tile + (lane & 15) * LDST + ((lane >> 4) << 3);
  half8 lo = *(const half8*)p;
  half8 hi = *(const half8*)(p + 16);
  v16h f;
#pragma unroll
  for (int i = 0; i < 8; ++i) { f[i] = lo[i]; f[i + 8] = hi[i]; }
  return f;
}

__device__ __forceinline__ v8f wmma16(v16h a, v16h b, v8f c) {
  return __builtin_amdgcn_wmma_f32_16x16x32_f16(false, a, false, b, (short)0, c,
                                                false, false);
}

// ---------------------------------------------------------------------------
// Tiled WMMA GEMM:  Out[M,N] = epi(A[M,K] @ Bt[N,K]^T + bias)
//   EPI 0: relu -> f16   EPI 1: linear -> f16   (require N % 128 == 0)
//   EPI 2: tanh -> f32 NT-store + sum((out-xref)^2) into *accum
// Block tile 128x128 (8 waves: 4 over M, 2 over N; wave tile 32x64), BK=32.
// Double-buffered LDS; f16 tiles staged by the Tensor Data Mover. After the
// K-loop the stage buffers are dead and are reused as a 128x136-half output
// tile so f16 results leave as coalesced b128 stores.
// Requires: M % 128 == 0, K % 16 == 0 (true for 784/512/256).
// ---------------------------------------------------------------------------
template <int EPI, bool AF32>
__global__ __launch_bounds__(256, 2)
void gemm_wmma(const void* __restrict__ A_, const _Float16* __restrict__ Bt,
               const float* __restrict__ bias, void* __restrict__ out_,
               const float* __restrict__ xref, float* __restrict__ accum,
               int M, int N, int K) {
  __shared__ alignas(16) _Float16 smem[4 * 128 * LDST];  // 40960 B
  _Float16* const Asb = smem;                  // [2][128*LDST]
  _Float16* const Bsb = smem + 2 * 128 * LDST; // [2][128*LDST]

  const int tid   = threadIdx.x;
  const int lane  = tid & 31;
  const int wave  = tid >> 5;
  const int wm    = wave & 3;
  const int wn    = wave >> 2;
  const int wRow  = wm * 32;
  const int wCol  = wn * 64;
  const int mBase = blockIdx.y * 128;
  const int nBase = blockIdx.x * 128;

  const int sRow = tid >> 1;         // 0..127
  const int sK   = (tid & 1) * 16;   // 0 or 16

  v8f acc[2][4];
#pragma unroll
  for (int i = 0; i < 2; ++i)
#pragma unroll
    for (int j = 0; j < 4; ++j) acc[i][j] = (v8f)0.0f;

  auto stage = [&](int kt, int p) {
    const int k0 = kt * 32;
    // ---- A tile ----
    if (AF32) {  // f32 input image: VALU convert, chunk-granularity guard
      _Float16* dst = Asb + p * 128 * LDST + sRow * LDST + sK;
      if (k0 + sK + 16 <= K) {
        const float* src = (const float*)A_ + (size_t)(mBase + sRow) * K + k0 + sK;
#pragma unroll
        for (int c = 0; c < 2; ++c) {
          f32x4 v0 = *(const f32x4*)(src + c * 8);
          f32x4 v1 = *(const f32x4*)(src + c * 8 + 4);
          half8 h;
          h[0] = (_Float16)v0[0]; h[1] = (_Float16)v0[1];
          h[2] = (_Float16)v0[2]; h[3] = (_Float16)v0[3];
          h[4] = (_Float16)v1[0]; h[5] = (_Float16)v1[1];
          h[6] = (_Float16)v1[2]; h[7] = (_Float16)v1[3];
          *(half8*)(dst + c * 8) = h;
        }
      } else {
        const half8 z = (half8)(_Float16)0.0f;
        *(half8*)dst = z;
        *(half8*)(dst + 8) = z;
      }
    } else {
#if USE_TDM
      if (wave == 0)
        tdm_load_tile_f16((uint32_t)(uintptr_t)(Asb + p * 128 * LDST),
                          (const _Float16*)A_ + (size_t)mBase * K + k0,
                          K - k0, M - mBase, 128, K);
#else
      _Float16* dst = Asb + p * 128 * LDST + sRow * LDST + sK;
      if (k0 + sK + 16 <= K) {
        const _Float16* src =
            (const _Float16*)A_ + (size_t)(mBase + sRow) * K + k0 + sK;
        *(half8*)dst = *(const half8*)src;
        *(half8*)(dst + 8) = *(const half8*)(src + 8);
      } else {
        const half8 z = (half8)(_Float16)0.0f;
        *(half8*)dst = z;
        *(half8*)(dst + 8) = z;
      }
#endif
    }
    // ---- B tile (Bt stored [N][K]) ----
#if USE_TDM
    if (wave == 0)
      tdm_load_tile_f16((uint32_t)(uintptr_t)(Bsb + p * 128 * LDST),
                        Bt + (size_t)nBase * K + k0, K - k0, N - nBase, 128, K);
#else
    {
      const int gN = nBase + sRow;
      _Float16* dst = Bsb + p * 128 * LDST + sRow * LDST + sK;
      if (gN < N && k0 + sK + 16 <= K) {
        const _Float16* src = Bt + (size_t)gN * K + k0 + sK;
        *(half8*)dst = *(const half8*)src;
        *(half8*)(dst + 8) = *(const half8*)(src + 8);
      } else {
        const half8 z = (half8)(_Float16)0.0f;
        *(half8*)dst = z;
        *(half8*)(dst + 8) = z;
      }
    }
#endif
  };

  auto compute = [&](int p) {
    v16h a[2], b[4];
#pragma unroll
    for (int i = 0; i < 2; ++i)
      a[i] = ld_frag(Asb + p * 128 * LDST + (wRow + i * 16) * LDST);
#pragma unroll
    for (int j = 0; j < 4; ++j)
      b[j] = ld_frag(Bsb + p * 128 * LDST + (wCol + j * 16) * LDST);
#pragma unroll
    for (int i = 0; i < 2; ++i)
#pragma unroll
      for (int j = 0; j < 4; ++j) acc[i][j] = wmma16(a[i], b[j], acc[i][j]);
  };

  const int nKT = (K + 31) >> 5;
  stage(0, 0);
  pipe_barrier();
  for (int kt = 0; kt < nKT; ++kt) {
    const int p = kt & 1;
    if (kt + 1 < nKT) stage(kt + 1, p ^ 1);
    compute(p);
    pipe_barrier();
  }

  // ---- epilogue (C/D layout: lanes 0-15 -> M+0..7, lanes 16-31 -> M+8..15) ----
  const int cLane = lane & 15;
  const int rHalf = (lane >> 4) * 8;
  if (EPI == 2) {
    float lsum = 0.0f;
#pragma unroll
    for (int i = 0; i < 2; ++i) {
#pragma unroll
      for (int j = 0; j < 4; ++j) {
        const int col = nBase + wCol + j * 16 + cLane;
        if (col < N) {
          const float bv = bias[col];
#pragma unroll
          for (int v = 0; v < 8; ++v) {
            const int row = mBase + wRow + i * 16 + rHalf + v;
            float val = tanhf(acc[i][j][v] + bv);
            const float xr =
                __builtin_nontemporal_load(&xref[(size_t)row * N + col]);
            const float d = val - xr;
            lsum += d * d;
            __builtin_nontemporal_store(val,
                                        &((float*)out_)[(size_t)row * N + col]);
          }
        }
      }
    }
#pragma unroll
    for (int m = 16; m >= 1; m >>= 1) lsum += __shfl_xor(lsum, m, 32);
    if (lane == 0) atomicAdd(accum, lsum);
  } else {
    // f16 output: deposit into the (now dead) stage LDS, then coalesced b128s.
    _Float16* const OutS = smem;  // 128 x OST halves = 34816 B <= 40960 B
#pragma unroll
    for (int i = 0; i < 2; ++i) {
#pragma unroll
      for (int j = 0; j < 4; ++j) {
        const int cl = wCol + j * 16 + cLane;
        const float bv = bias[nBase + cl];
#pragma unroll
        for (int v = 0; v < 8; ++v) {
          float val = acc[i][j][v] + bv;
          if (EPI == 0) val = val > 0.0f ? val : 0.0f;
          OutS[(wRow + i * 16 + rHalf + v) * OST + cl] = (_Float16)val;
        }
      }
    }
    __syncthreads();
#pragma unroll
    for (int pass = 0; pass < 4; ++pass) {
      const int r = pass * 32 + (tid >> 3);
      const int c = (tid & 7) * 16;
      const _Float16* s = &OutS[r * OST + c];
      _Float16* g = (_Float16*)out_ + (size_t)(mBase + r) * N + nBase + c;
      *(half8*)g       = *(const half8*)s;
      *(half8*)(g + 8) = *(const half8*)(s + 8);
    }
  }
}

// ---------------------------------------------------------------------------
// Residual VQ: one block = 64 rows vs all 256 codes; scores via WMMA (K=256);
// argmin ||r-c||^2 == argmax(r.c - 0.5||c||^2).
// Pass 1: Rout = r - cb[idx] (residual). Pass 2: Rout = enc - (r1 - cb[idx])
// = quantized (in place). Commit partials via f32 atomics.
// ---------------------------------------------------------------------------
template <bool SECOND>
__global__ __launch_bounds__(256, 2)
void vq_kernel(const _Float16* __restrict__ R, const _Float16* __restrict__ cb,
               const float* __restrict__ cnorm, const _Float16* __restrict__ enc,
               _Float16* __restrict__ Rout, int* __restrict__ idx_out,
               float* __restrict__ commit_acc) {
  __shared__ alignas(16) _Float16 As[2][64 * LDST];
  __shared__ alignas(16) _Float16 Bs[2][256 * LDST];
  __shared__ float redV[4][64];
  __shared__ int   redC[4][64];
  __shared__ int   bestIdx[64];

  const int tid  = threadIdx.x;
  const int lane = tid & 31;
  const int wave = tid >> 5;
  const int wm   = wave & 1;     // 2 waves over 64 rows
  const int wn   = wave >> 1;    // 4 waves over 256 codes
  const int wRow = wm * 32;
  const int wCol = wn * 64;
  const size_t rowBase = (size_t)blockIdx.x * 64;

  v8f acc[2][4];
#pragma unroll
  for (int i = 0; i < 2; ++i)
#pragma unroll
    for (int j = 0; j < 4; ++j) acc[i][j] = (v8f)0.0f;

  auto stage = [&](int kt, int p) {
    const int k0 = kt * 32;
#if USE_TDM
    if (wave == 0) {
      tdm_load_tile_f16((uint32_t)(uintptr_t)&As[p][0], R + rowBase * 256 + k0,
                        256 - k0, 64, 64, 256);
      tdm_load_tile_f16((uint32_t)(uintptr_t)&Bs[p][0], cb + k0,
                        256 - k0, 256, 256, 256);
    }
#else
    {
      const int r  = tid >> 2;
      const int kb = (tid & 3) * 8;
      *(half8*)&As[p][r * LDST + kb] =
          *(const half8*)(R + (rowBase + r) * 256 + k0 + kb);
      const _Float16* src = cb + (size_t)tid * 256 + k0;
#pragma unroll
      for (int b2 = 0; b2 < 32; b2 += 8)
        *(half8*)&Bs[p][tid * LDST + b2] = *(const half8*)(src + b2);
    }
#endif
  };

  auto compute = [&](int p) {
    v16h a[2], b[4];
#pragma unroll
    for (int i = 0; i < 2; ++i) a[i] = ld_frag(&As[p][(wRow + i * 16) * LDST]);
#pragma unroll
    for (int j = 0; j < 4; ++j) b[j] = ld_frag(&Bs[p][(wCol + j * 16) * LDST]);
#pragma unroll
    for (int i = 0; i < 2; ++i)
#pragma unroll
      for (int j = 0; j < 4; ++j) acc[i][j] = wmma16(a[i], b[j], acc[i][j]);
  };

  stage(0, 0);
  pipe_barrier();
  for (int kt = 0; kt < 8; ++kt) {      // K = 256
    const int p = kt & 1;
    if (kt + 1 < 8) stage(kt + 1, p ^ 1);
    compute(p);
    pipe_barrier();
  }

  // ---- per-row argmax of (dot - 0.5*||c||^2) ----
  const int cLane = lane & 15;
  const int half  = lane >> 4;
#pragma unroll
  for (int i = 0; i < 2; ++i) {
#pragma unroll
    for (int v = 0; v < 8; ++v) {
      float bv = -1e30f; int bc = 0;
#pragma unroll
      for (int j = 0; j < 4; ++j) {
        const int col = wCol + j * 16 + cLane;
        const float s = acc[i][j][v] - 0.5f * cnorm[col];
        if (s > bv || (s == bv && col < bc)) { bv = s; bc = col; }
      }
#pragma unroll
      for (int m = 1; m < 16; m <<= 1) {   // xor masks < 16 stay within half
        const float ov = __shfl_xor(bv, m, 32);
        const int   oc = __shfl_xor(bc, m, 32);
        if (ov > bv || (ov == bv && oc < bc)) { bv = ov; bc = oc; }
      }
      if (cLane == 0) {
        const int rl = wRow + i * 16 + half * 8 + v;
        redV[wn][rl] = bv;
        redC[wn][rl] = bc;
      }
    }
  }
  __syncthreads();
  if (tid < 64) {
    float bv = redV[0][tid]; int bc = redC[0][tid];
#pragma unroll
    for (int w = 1; w < 4; ++w) {
      const float ov = redV[w][tid]; const int oc = redC[w][tid];
      if (ov > bv || (ov == bv && oc < bc)) { bv = ov; bc = oc; }
    }
    bestIdx[tid] = bc;
    idx_out[2 * (rowBase + tid) + (SECOND ? 1 : 0)] = bc;
  }
  __syncthreads();

  // ---- residual / quantized + commit partial (4 threads per row) ----
  {
    const int r    = tid >> 2;
    const int db   = (tid & 3) * 64;
    const int code = bestIdx[r];
    const size_t g = (rowBase + r) * 256 + db;
    float sq = 0.0f;
#pragma unroll 8
    for (int d = 0; d < 64; ++d) {
      const float rv  = (float)R[g + d];
      const float cv  = (float)cb[(size_t)code * 256 + db + d];
      const float res = rv - cv;
      sq += res * res;
      const float outv = SECOND ? ((float)enc[g + d] - res) : res;
      Rout[g + d] = (_Float16)outv;
    }
#pragma unroll
    for (int m = 16; m >= 1; m >>= 1) sq += __shfl_xor(sq, m, 32);
    if (lane == 0) atomicAdd(commit_acc, sq);
  }
}

// ---------------------------------------------------------------------------
// Small prep / finalize kernels
// ---------------------------------------------------------------------------
__global__ void transpose_to_f16(const float* __restrict__ W,
                                 _Float16* __restrict__ Wt, int K, int N) {
  const int n = blockIdx.x * 32 + (threadIdx.x & 31);
  const int k = blockIdx.y * 8 + (threadIdx.x >> 5);
  if (n < N && k < K) Wt[(size_t)n * K + k] = (_Float16)W[(size_t)k * N + n];
}

__global__ void cb_prep(const float* __restrict__ cb, _Float16* __restrict__ cbh,
                        float* __restrict__ cnorm) {
  const int c = threadIdx.x;  // 256 threads, one block
  float s = 0.0f;
  for (int d = 0; d < 256; ++d) {
    const float v = cb[c * 256 + d];
    cbh[c * 256 + d] = (_Float16)v;
    s += v * v;
  }
  cnorm[c] = s;
}

__global__ void zero_accums(float* a) {
  if (threadIdx.x < 4) a[threadIdx.x] = 0.0f;
}

__global__ void finalize_loss(const float* __restrict__ a,
                              float* __restrict__ loss) {
  *loss = a[0] / (65536.0f * 784.0f) + 0.25f * a[1] / (65536.0f * 256.0f);
}

// ---------------------------------------------------------------------------
// Host-side launch
// ---------------------------------------------------------------------------
extern "C" void kernel_launch(void* const* d_in, const int* in_sizes, int n_in,
                              void* d_out, int out_size, void* d_ws,
                              size_t ws_size, hipStream_t stream) {
  const float* x   = (const float*)d_in[0];
  const float* We1 = (const float*)d_in[1];
  const float* be1 = (const float*)d_in[2];
  const float* We2 = (const float*)d_in[3];
  const float* be2 = (const float*)d_in[4];
  const float* cb0 = (const float*)d_in[5];
  const float* cb1 = (const float*)d_in[6];
  const float* Wd1 = (const float*)d_in[7];
  const float* bd1 = (const float*)d_in[8];
  const float* Wd2 = (const float*)d_in[9];
  const float* bd2 = (const float*)d_in[10];

  constexpr int B = 65536, DIN = 784, DH = 512, DL = 256, NC = 256;

  char* ws = (char*)d_ws;
  size_t off = 0;
  auto alloc = [&](size_t bytes) -> void* {
    void* p = ws + off;
    off = (off + bytes + 255) & ~(size_t)255;
    return p;
  };
  _Float16* We1t = (_Float16*)alloc(sizeof(_Float16) * (size_t)DH * DIN);
  _Float16* We2t = (_Float16*)alloc(sizeof(_Float16) * (size_t)DL * DH);
  _Float16* Wd1t = (_Float16*)alloc(sizeof(_Float16) * (size_t)DH * DL);
  _Float16* Wd2t = (_Float16*)alloc(sizeof(_Float16) * (size_t)DIN * DH);
  _Float16* cb0h = (_Float16*)alloc(sizeof(_Float16) * (size_t)NC * DL);
  _Float16* cb1h = (_Float16*)alloc(sizeof(_Float16) * (size_t)NC * DL);
  float* cn0     = (float*)alloc(sizeof(float) * NC);
  float* cn1     = (float*)alloc(sizeof(float) * NC);
  float* accums  = (float*)alloc(sizeof(float) * 4);
  _Float16* h    = (_Float16*)alloc(sizeof(_Float16) * (size_t)B * DH);
  _Float16* enc  = (_Float16*)alloc(sizeof(_Float16) * (size_t)B * DL);
  _Float16* r1   = (_Float16*)alloc(sizeof(_Float16) * (size_t)B * DL);

  float* recon    = (float*)d_out;
  int*   idx_out  = (int*)((float*)d_out + (size_t)B * DIN);
  float* loss_out = (float*)d_out + (size_t)B * DIN + 2 * (size_t)B;

  zero_accums<<<1, 32, 0, stream>>>(accums);
  transpose_to_f16<<<dim3((DH + 31) / 32, (DIN + 7) / 8), 256, 0, stream>>>(We1, We1t, DIN, DH);
  transpose_to_f16<<<dim3((DL + 31) / 32, (DH + 7) / 8), 256, 0, stream>>>(We2, We2t, DH, DL);
  transpose_to_f16<<<dim3((DH + 31) / 32, (DL + 7) / 8), 256, 0, stream>>>(Wd1, Wd1t, DL, DH);
  transpose_to_f16<<<dim3((DIN + 31) / 32, (DH + 7) / 8), 256, 0, stream>>>(Wd2, Wd2t, DH, DIN);
  cb_prep<<<1, 256, 0, stream>>>(cb0, cb0h, cn0);
  cb_prep<<<1, 256, 0, stream>>>(cb1, cb1h, cn1);

  // encoder
  gemm_wmma<0, true><<<dim3(DH / 128, B / 128), 256, 0, stream>>>(
      x, We1t, be1, h, nullptr, nullptr, B, DH, DIN);
  gemm_wmma<1, false><<<dim3(DL / 128, B / 128), 256, 0, stream>>>(
      h, We2t, be2, enc, nullptr, nullptr, B, DL, DH);
  // residual VQ (two quantizers)
  vq_kernel<false><<<B / 64, 256, 0, stream>>>(enc, cb0h, cn0, nullptr, r1,
                                               idx_out, accums + 1);
  vq_kernel<true><<<B / 64, 256, 0, stream>>>(r1, cb1h, cn1, enc, r1, idx_out,
                                              accums + 1);
  // decoder (h buffer reused)
  gemm_wmma<0, false><<<dim3(DH / 128, B / 128), 256, 0, stream>>>(
      r1, Wd1t, bd1, h, nullptr, nullptr, B, DH, DL);
  gemm_wmma<2, false><<<dim3((DIN + 127) / 128, B / 128), 256, 0, stream>>>(
      h, Wd2t, bd2, recon, x, accums + 0, B, DIN, DH);

  finalize_loss<<<1, 1, 0, stream>>>(accums, loss_out);
}
